// UnifiedBPModel_45019847197278
// MI455X (gfx1250) — matrix operand order repaired and checked by reading, not compiled, not run on hardware
//
#include <hip/hip_runtime.h>
#include <hip/hip_bf16.h>

// ---------------------------------------------------------------------------
// UnifiedBPModel forward on MI455X (gfx1250, wave32, WMMA).
// All large GEMMs + attention matmuls run on v_wmma_f32_16x16x32_bf16
// (bf16 inputs, f32 accumulate). Staging is vectorized (b128 loads,
// packed bf16 cvt, ds_store_b64) and double-buffered in LDS so tile loads
// overlap WMMA. Small ragged aggregation / layernorms / elementwise run on
// VALU with LDS reductions.
// ---------------------------------------------------------------------------

typedef __attribute__((ext_vector_type(16))) __bf16 v16bf;
typedef __attribute__((ext_vector_type(4)))  __bf16 bf16x4;
typedef __attribute__((ext_vector_type(8)))  float  v8f;

#define D_MODEL 512
#define N_HEADS 8
#define HD      64
#define N_PATH  2048
#define L_SEQ   64
#define N_B     64
#define N_CAT   32
#define N_CLS   512

static __device__ __forceinline__ v8f wmma_bf16(v16bf a, v16bf b, v8f c) {
  // D = A(16x32) * B(32x16) + C, f32 accumulate
  return __builtin_amdgcn_wmma_f32_16x16x32_bf16(false, a, false, b,
                                                 (short)0, c, false, false);
}

// A fragment: 16x32 bf16, row-major source with leading dim `ld`.
// ISA layout: lanes 0-15 row m: V0..V3 = K 0..7 (pairs), V4..V7 = K 16..23;
//             lanes 16-31: K 8..15 and K 24..31.
static __device__ __forceinline__ v16bf frag_a_bf(const __bf16* src, int ld) {
  int lane = threadIdx.x & 31;
  int m = lane & 15, hi = lane >> 4;
  v16bf f;
#pragma unroll
  for (int r = 0; r < 8; ++r) {
    int k = 2 * (r & 3) + ((r & 4) ? 16 : 0) + hi * 8;
    f[2 * r]     = src[m * ld + k];
    f[2 * r + 1] = src[m * ld + k + 1];
  }
  return f;
}

// Same A layout but gathering from f32 source (converted on the fly).
static __device__ __forceinline__ v16bf frag_a_f32(const float* src, int ld) {
  int lane = threadIdx.x & 31;
  int m = lane & 15, hi = lane >> 4;
  v16bf f;
#pragma unroll
  for (int r = 0; r < 8; ++r) {
    int k = 2 * (r & 3) + ((r & 4) ? 16 : 0) + hi * 8;
    f[2 * r]     = (__bf16)src[m * ld + k];
    f[2 * r + 1] = (__bf16)src[m * ld + k + 1];
  }
  return f;
}

// B fragment 32x16 from [N,K] storage (weight layout W[n][k], ld = K stride).
// Lanes 0-15 col n: K 0..7, 16..23 in V0..V7 pairs; lanes 16-31: K 8..15, 24..31.
static __device__ __forceinline__ v16bf frag_b_nk(const __bf16* src, int ld) {
  int lane = threadIdx.x & 31;
  int n = lane & 15, hi = lane >> 4;
  v16bf f;
#pragma unroll
  for (int r = 0; r < 8; ++r) {
    int k = 2 * (r & 3) + ((r & 4) ? 16 : 0) + hi * 8;
    f[2 * r]     = src[n * ld + k];
    f[2 * r + 1] = src[n * ld + k + 1];
  }
  return f;
}

// B fragment 32x16 from [K,N] storage (e.g. V matrix), ld = N stride.
static __device__ __forceinline__ v16bf frag_b_kn(const __bf16* src, int ld) {
  int lane = threadIdx.x & 31;
  int n = lane & 15, hi = lane >> 4;
  v16bf f;
#pragma unroll
  for (int r = 0; r < 8; ++r) {
    int k = 2 * (r & 3) + ((r & 4) ? 16 : 0) + hi * 8;
    f[2 * r]     = src[k * ld + n];
    f[2 * r + 1] = src[(k + 1) * ld + n];
  }
  return f;
}

static __device__ __forceinline__ bf16x4 cvt_bf16x4(float4 v) {
  bf16x4 o;
  o.x = (__bf16)v.x; o.y = (__bf16)v.y; o.z = (__bf16)v.z; o.w = (__bf16)v.w;
  return o;
}

// ---------------------------------------------------------------------------
// Generic tiled WMMA GEMM:  C[M,N] = act( A[M,K] * B + bias + resid )
//   b_kn == 0 : B given as W[N,K] (PyTorch x @ W.T)
//   b_kn == 1 : B given as W[K,N]
//   act: 0 = none, 1 = relu, 2 = sigmoid
// Block = 256 threads (8 waves), tile 128x128, K step 32, K % 32 == 0.
// ALIGNED instantiation (M%128==0, N%128==0, b_kn==0): no bounds checks,
// vectorized staging, LDS double buffering, prefetch 2 tiles ahead.
// ---------------------------------------------------------------------------
template <bool ALIGNED>
__global__ __launch_bounds__(256) void gemm_wmma_kernel(
    const float* __restrict__ A, const float* __restrict__ W,
    const float* bias, const float* resid, float* C,
    int M, int N, int K, int act, int b_kn)
{
  __shared__ __bf16 sA[2][128 * 32];
  __shared__ __bf16 sB[2][128 * 32];   // stored [n][k]

  int tid   = threadIdx.x;
  int tileM = blockIdx.x * 128;
  int tileN = blockIdx.y * 128;
  int wave  = tid >> 5;
  int wm    = wave & 1;   // 2 M groups of 64 rows
  int wn    = wave >> 1;  // 4 N groups of 32 cols

  v8f zero = {0.f, 0.f, 0.f, 0.f, 0.f, 0.f, 0.f, 0.f};
  v8f acc[4][2];
#pragma unroll
  for (int i = 0; i < 4; ++i)
#pragma unroll
    for (int j = 0; j < 2; ++j) acc[i][j] = zero;

  // ---- tile staging ----
  auto stage = [&](int buf, int k0) {
    if (ALIGNED) {
      // 1024 float4 per matrix; thread t handles 4, coalesced 16B chunks.
#pragma unroll
      for (int j = 0; j < 4; ++j) {
        int idx = tid + 256 * j;
        int m = idx >> 3, q = (idx & 7) << 2;        // row, float4 col*4
        float4 av = *reinterpret_cast<const float4*>(
            &A[(long long)(tileM + m) * K + k0 + q]);
        *reinterpret_cast<bf16x4*>(&sA[buf][m * 32 + q]) = cvt_bf16x4(av);
        float4 bv = *reinterpret_cast<const float4*>(
            &W[(long long)(tileN + m) * K + k0 + q]);
        *reinterpret_cast<bf16x4*>(&sB[buf][m * 32 + q]) = cvt_bf16x4(bv);
      }
    } else {
      for (int i = tid; i < 128 * 32; i += 256) {
        int m  = i >> 5;
        int kk = i & 31;
        int gm = tileM + m;
        int gk = k0 + kk;
        float va = (gm < M) ? A[(long long)gm * K + gk] : 0.f;
        sA[buf][i] = (__bf16)va;
        int gn = tileN + m;
        float vb = 0.f;
        if (gn < N)
          vb = b_kn ? W[(long long)gk * N + gn] : W[(long long)gn * K + gk];
        sB[buf][i] = (__bf16)vb;
      }
    }
  };

  int nk = K >> 5;
  stage(0, 0);
  for (int kt = 0; kt < nk; ++kt) {
    __syncthreads();                       // tile kt staged for everyone
    if (kt + 1 < nk) stage((kt + 1) & 1, (kt + 1) << 5);
    if (ALIGNED && kt + 2 < nk) {
      // gfx1250 global_prefetch_b8: pull K-tile kt+2 toward L2/WGP$
      int m = tid >> 1, half = (tid & 1) << 4;
      __builtin_prefetch(&A[(long long)(tileM + m) * K + ((kt + 2) << 5) + half], 0, 1);
      __builtin_prefetch(&W[(long long)(tileN + m) * K + ((kt + 2) << 5) + half], 0, 1);
    }
    int buf = kt & 1;
    v16bf afrag[4];
#pragma unroll
    for (int mi = 0; mi < 4; ++mi)
      afrag[mi] = frag_a_bf(&sA[buf][(wm * 64 + mi * 16) * 32], 32);
#pragma unroll
    for (int ni = 0; ni < 2; ++ni) {
      v16bf bfr = frag_b_nk(&sB[buf][(wn * 32 + ni * 16) * 32], 32);
#pragma unroll
      for (int mi = 0; mi < 4; ++mi)
        acc[mi][ni] = wmma_bf16(afrag[mi], bfr, acc[mi][ni]);
    }
  }

  // epilogue: C/D layout — lane (hi,n), VGPR r -> row hi*8+r, col n
  int lane = tid & 31;
  int nlo = lane & 15, hi2 = lane >> 4;
#pragma unroll
  for (int mi = 0; mi < 4; ++mi)
#pragma unroll
    for (int ni = 0; ni < 2; ++ni)
#pragma unroll
      for (int r = 0; r < 8; ++r) {
        int row = tileM + wm * 64 + mi * 16 + hi2 * 8 + r;
        int col = tileN + wn * 32 + ni * 16 + nlo;
        if (ALIGNED || (row < M && col < N)) {
          float v = acc[mi][ni][r];
          if (bias)  v += bias[col];
          if (resid) v += resid[(long long)row * N + col];
          if (act == 1)      v = fmaxf(v, 0.f);
          else if (act == 2) v = 1.f / (1.f + __expf(-v));
          C[(long long)row * N + col] = v;
        }
      }
}

// ---------------------------------------------------------------------------
// Fused per-(path, head) self-attention: softmax((q k^T)*s + mask) v
// qkv chunk: [pchunk*64, 1536] rows; one block per (local_path, head).
// ---------------------------------------------------------------------------
__global__ __launch_bounds__(256) void path_attn_kernel(
    const float* __restrict__ qkv, const int* __restrict__ masks,
    float* __restrict__ out, int p0)
{
  int lp   = blockIdx.x >> 3;   // path within chunk
  int head = blockIdx.x & 7;
  int p    = p0 + lp;           // global path

  __shared__ __bf16 sQ[64 * 64];
  __shared__ __bf16 sK[64 * 64];
  __shared__ __bf16 sV[64 * 64];
  __shared__ float  sL[64 * 64];

  int tid = threadIdx.x;
  // vectorized q/k/v staging: 1024 float4 per matrix, 4 per thread
#pragma unroll
  for (int j = 0; j < 4; ++j) {
    int idx = tid + 256 * j;
    int l = idx >> 4, q = (idx & 15) << 2;
    const float* row = qkv + (long long)(lp * 64 + l) * 1536 + head * 64 + q;
    *reinterpret_cast<bf16x4*>(&sQ[l * 64 + q]) =
        cvt_bf16x4(*reinterpret_cast<const float4*>(row));
    *reinterpret_cast<bf16x4*>(&sK[l * 64 + q]) =
        cvt_bf16x4(*reinterpret_cast<const float4*>(row + 512));
    *reinterpret_cast<bf16x4*>(&sV[l * 64 + q]) =
        cvt_bf16x4(*reinterpret_cast<const float4*>(row + 1024));
  }
  __syncthreads();

  int wave = tid >> 5, lane = tid & 31;
  int lt = wave >> 1;   // 4 row groups of 16
  int nt = wave & 1;    // 2 col groups of 32
  int nlo = lane & 15, hi = lane >> 4;

  v8f zero = {0.f, 0.f, 0.f, 0.f, 0.f, 0.f, 0.f, 0.f};

  // logits = q k^T : A = sQ rows, B = sK in [N,K] layout (k[m][d])
  v8f acc[2] = {zero, zero};
#pragma unroll
  for (int k0 = 0; k0 < 64; k0 += 32) {
    v16bf a = frag_a_bf(sQ + lt * 16 * 64 + k0, 64);
#pragma unroll
    for (int ni = 0; ni < 2; ++ni) {
      v16bf b = frag_b_nk(sK + (nt * 32 + ni * 16) * 64 + k0, 64);
      acc[ni] = wmma_bf16(a, b, acc[ni]);
    }
  }
#pragma unroll
  for (int ni = 0; ni < 2; ++ni)
#pragma unroll
    for (int r = 0; r < 8; ++r) {
      int l    = lt * 16 + hi * 8 + r;
      int mcol = nt * 32 + ni * 16 + nlo;
      float lg = acc[ni][r] * 0.125f;                   // 1/sqrt(64)
      if (masks[(long long)p * 64 + mcol] == 0) lg = -1e9f;  // key-pad mask
      sL[l * 64 + mcol] = lg;
    }
  __syncthreads();

  // row softmax (64 rows, one thread each)
  if (tid < 64) {
    float* row = sL + tid * 64;
    float mx = -3.4e38f;
    for (int j = 0; j < 64; ++j) mx = fmaxf(mx, row[j]);
    float s = 0.f;
    for (int j = 0; j < 64; ++j) { float e = __expf(row[j] - mx); row[j] = e; s += e; }
    float inv = 1.f / s;
    for (int j = 0; j < 64; ++j) row[j] *= inv;
  }
  __syncthreads();

  // o = a v : A = sL (f32->bf16), B = sV in [K,N] layout (v[m][dd])
  v8f oacc[2] = {zero, zero};
#pragma unroll
  for (int k0 = 0; k0 < 64; k0 += 32) {
    v16bf a = frag_a_f32(sL + lt * 16 * 64 + k0, 64);
#pragma unroll
    for (int ni = 0; ni < 2; ++ni) {
      v16bf b = frag_b_kn(sV + k0 * 64 + nt * 32 + ni * 16, 64);
      oacc[ni] = wmma_bf16(a, b, oacc[ni]);
    }
  }
#pragma unroll
  for (int ni = 0; ni < 2; ++ni)
#pragma unroll
    for (int r = 0; r < 8; ++r) {
      int l  = lt * 16 + hi * 8 + r;
      int dd = nt * 32 + ni * 16 + nlo;
      out[(long long)(p * 64 + l) * 512 + head * 64 + dd] = oacc[ni][r];
    }
}

// ---------------------------------------------------------------------------
// Ragged cross-attention aggregation: one block per (sample b, head h).
// logits over P paths masked by path_to_sample[p]==b; softmax; weighted V sum.
// ---------------------------------------------------------------------------
__global__ __launch_bounds__(256) void agg_attn_kernel(
    const float* __restrict__ q, const float* __restrict__ k,
    const float* __restrict__ v, const int* __restrict__ p2s,
    float* __restrict__ out)
{
  int b = blockIdx.x >> 3;
  int h = blockIdx.x & 7;
  __shared__ float sq[64];
  __shared__ float sLog[2048];
  __shared__ float sRed[256];
  __shared__ float sAcc[64];
  int tid = threadIdx.x;
  if (tid < 64) { sq[tid] = q[b * 512 + h * 64 + tid]; sAcc[tid] = 0.f; }
  __syncthreads();

  float lmax = -3.4e38f;
  for (int p = tid; p < 2048; p += 256) {
    float lg = -1e9f;
    if (p2s[p] == b) {
      const float* kp = k + (long long)p * 512 + h * 64;
      float dot = 0.f;
#pragma unroll
      for (int d = 0; d < 64; ++d) dot += sq[d] * kp[d];
      lg = dot * 0.125f;
    }
    sLog[p] = lg;
    lmax = fmaxf(lmax, lg);
  }
  sRed[tid] = lmax;
  __syncthreads();
  for (int s = 128; s > 0; s >>= 1) {
    if (tid < s) sRed[tid] = fmaxf(sRed[tid], sRed[tid + s]);
    __syncthreads();
  }
  float mx = sRed[0];
  __syncthreads();

  float lsum = 0.f;
  float racc[64];
#pragma unroll
  for (int d = 0; d < 64; ++d) racc[d] = 0.f;
  for (int p = tid; p < 2048; p += 256) {
    float w = __expf(sLog[p] - mx);
    lsum += w;
    const float* vp = v + (long long)p * 512 + h * 64;
#pragma unroll
    for (int d = 0; d < 64; ++d) racc[d] += w * vp[d];
  }
  sRed[tid] = lsum;
  __syncthreads();
  for (int s = 128; s > 0; s >>= 1) {
    if (tid < s) sRed[tid] += sRed[tid + s];
    __syncthreads();
  }
#pragma unroll
  for (int d = 0; d < 64; ++d) atomicAdd(&sAcc[d], racc[d]);  // ds_add_f32
  __syncthreads();
  if (tid < 64) out[b * 512 + h * 64 + tid] = sAcc[tid] / sRed[0];
}

// ---------------------------------------------------------------------------
// LayerNorm over rows of length 512 (all norms in this model are over D=512).
// One block per row; act: 0 none, 1 relu. Supports strided input rows.
// ---------------------------------------------------------------------------
__global__ __launch_bounds__(256) void layernorm512_kernel(
    const float* __restrict__ in, long long in_stride,
    const float* __restrict__ w, const float* __restrict__ b,
    float* __restrict__ out, long long out_stride, int act)
{
  __shared__ float red[256];
  long long row = blockIdx.x;
  int tid = threadIdx.x;
  float x0 = in[row * in_stride + tid];
  float x1 = in[row * in_stride + tid + 256];
  red[tid] = x0 + x1;
  __syncthreads();
  for (int s = 128; s > 0; s >>= 1) { if (tid < s) red[tid] += red[tid + s]; __syncthreads(); }
  float mu = red[0] * (1.f / 512.f);
  __syncthreads();
  float d0 = x0 - mu, d1 = x1 - mu;
  red[tid] = d0 * d0 + d1 * d1;
  __syncthreads();
  for (int s = 128; s > 0; s >>= 1) { if (tid < s) red[tid] += red[tid + s]; __syncthreads(); }
  float rstd = rsqrtf(red[0] * (1.f / 512.f) + 1e-5f);
  float y0 = d0 * rstd * w[tid] + b[tid];
  float y1 = d1 * rstd * w[tid + 256] + b[tid + 256];
  if (act == 1) { y0 = fmaxf(y0, 0.f); y1 = fmaxf(y1, 0.f); }
  out[row * out_stride + tid]       = y0;
  out[row * out_stride + tid + 256] = y1;
}

// ---------------------------------------------------------------------------
// Elementwise helpers
// ---------------------------------------------------------------------------
__global__ void add_pe_kernel(const float* __restrict__ paths,
                              float* __restrict__ h, long long n)
{
  long long i = (long long)blockIdx.x * blockDim.x + threadIdx.x;
  if (i >= n) return;
  int c = (int)(i & 511);
  int l = (int)((i >> 9) & 63);
  // div = exp((c & ~1) * (-ln(10000)/512)); even -> sin, odd -> cos
  float freq = __expf((float)(c & ~1) * (-9.210340371976184f / 512.f));
  float ang  = (float)l * freq;
  float pe   = (c & 1) ? __cosf(ang) : __sinf(ang);
  h[i] = paths[i] + pe;
}

__global__ void sigmoid_kernel(const float* __restrict__ in,
                               float* __restrict__ out, int n)
{
  int i = blockIdx.x * blockDim.x + threadIdx.x;
  if (i < n) out[i] = 1.f / (1.f + __expf(-in[i]));
}

__global__ void select_features_kernel(const float* __restrict__ aggn,
                                       const float* __restrict__ venh,
                                       const int* __restrict__ pps,
                                       float* __restrict__ feat, int n)
{
  int i = blockIdx.x * blockDim.x + threadIdx.x;
  if (i >= n) return;
  int row = i >> 9;
  feat[i] = (pps[row] > 0) ? aggn[i] : venh[i];
}

__global__ void build_comb_kernel(const float* __restrict__ feat,
                                  const float* __restrict__ g,
                                  const float* __restrict__ wce,
                                  float* __restrict__ comb, int n)
{
  int i = blockIdx.x * blockDim.x + threadIdx.x;
  if (i >= n) return;
  int row = i >> 9, c = i & 511;
  float f = feat[i];
  comb[row * 1024 + c]       = f;                     // features
  comb[row * 1024 + 512 + c] = f + g[i] * wce[i];     // enhanced
}

// ---------------------------------------------------------------------------
// Host orchestration
// ---------------------------------------------------------------------------
extern "C" void kernel_launch(void* const* d_in, const int* in_sizes, int n_in,
                              void* d_out, int out_size, void* d_ws, size_t ws_size,
                              hipStream_t stream)
{
  (void)in_sizes; (void)n_in; (void)out_size;

  const float* variants = (const float*)d_in[0];   // [64, 512]
  const float* paths    = (const float*)d_in[1];   // [2048, 64, 512]
  const int*   masks    = (const int*)d_in[2];     // [2048, 64] bool->int
  const int*   p2s      = (const int*)d_in[3];     // [2048]
  const int*   pps      = (const int*)d_in[4];     // [64]
  int pi = 5;
  const float* enc_qkv_w = (const float*)d_in[pi++];  // [2,1536,512]
  const float* enc_qkv_b = (const float*)d_in[pi++];  // [2,1536]
  const float* enc_out_w = (const float*)d_in[pi++];  // [2,512,512]
  const float* enc_out_b = (const float*)d_in[pi++];
  const float* enc_ln1_w = (const float*)d_in[pi++];
  const float* enc_ln1_b = (const float*)d_in[pi++];
  const float* enc_ln2_w = (const float*)d_in[pi++];
  const float* enc_ln2_b = (const float*)d_in[pi++];
  const float* enc_ff1_w = (const float*)d_in[pi++];  // [2,2048,512]
  const float* enc_ff1_b = (const float*)d_in[pi++];
  const float* enc_ff2_w = (const float*)d_in[pi++];  // [2,512,2048]
  const float* enc_ff2_b = (const float*)d_in[pi++];
  const float* path_norm_w = (const float*)d_in[pi++];
  const float* path_norm_b = (const float*)d_in[pi++];
  const float* vp_w    = (const float*)d_in[pi++];    // [512,512]
  const float* vp_b    = (const float*)d_in[pi++];
  const float* vp_ln_w = (const float*)d_in[pi++];
  const float* vp_ln_b = (const float*)d_in[pi++];
  const float* agg_qkv_w = (const float*)d_in[pi++];  // [1536,512]
  const float* agg_qkv_b = (const float*)d_in[pi++];
  const float* agg_out_w = (const float*)d_in[pi++];  // [512,512]
  const float* agg_out_b = (const float*)d_in[pi++];
  const float* agg_norm_w = (const float*)d_in[pi++];
  const float* agg_norm_b = (const float*)d_in[pi++];
  const float* cat_w1 = (const float*)d_in[pi++];     // [512,512]
  const float* cat_b1 = (const float*)d_in[pi++];
  const float* cat_ln_w = (const float*)d_in[pi++];
  const float* cat_ln_b = (const float*)d_in[pi++];
  const float* cat_w2 = (const float*)d_in[pi++];     // [32,512]
  const float* cat_b2 = (const float*)d_in[pi++];
  const float* cat_emb = (const float*)d_in[pi++];    // [32,512]  ([K,N])
  const float* gate_w1 = (const float*)d_in[pi++];    // [512,32]
  const float* gate_b1 = (const float*)d_in[pi++];
  const float* gate_w2 = (const float*)d_in[pi++];    // [512,512]
  const float* gate_b2 = (const float*)d_in[pi++];
  const float* lab_w1 = (const float*)d_in[pi++];     // [512,1024]
  const float* lab_b1 = (const float*)d_in[pi++];
  const float* lab_ln_w = (const float*)d_in[pi++];
  const float* lab_ln_b = (const float*)d_in[pi++];
  const float* lab_w2 = (const float*)d_in[pi++];     // [256,512]
  const float* lab_b2 = (const float*)d_in[pi++];
  const float* lab_w3 = (const float*)d_in[pi++];     // [512,256]
  const float* lab_b3 = (const float*)d_in[pi++];

  // ---- workspace layout (floats) ----
  float* ws = (float*)d_ws;
  size_t off = 0;
  auto wsalloc = [&](size_t n) { float* p = ws + off; off += n; return p; };
  const size_t MTOK = (size_t)N_PATH * L_SEQ;          // 131072 rows
  float* h    = wsalloc(MTOK * 512);                   // 256 MB
  float* xbuf = wsalloc(MTOK * 512);                   // 256 MB
  float* big  = wsalloc((size_t)32768 * 2048);         // 256 MB (chunk scratch)
  float* path_emb = wsalloc((size_t)N_PATH * 512);
  float* kagg = wsalloc((size_t)N_PATH * 512);
  float* vagg = wsalloc((size_t)N_PATH * 512);
  float* venh = wsalloc((size_t)N_B * 512);
  float* qagg = wsalloc((size_t)N_B * 512);
  float* aggheads = wsalloc((size_t)N_B * 512);
  float* aggn = wsalloc((size_t)N_B * 512);
  float* tmp1 = wsalloc((size_t)N_B * 512);
  float* wce  = wsalloc((size_t)N_B * 512);
  float* ghid = wsalloc((size_t)N_B * 512);
  float* gbuf = wsalloc((size_t)N_B * 512);
  float* lab1 = wsalloc((size_t)N_B * 512);
  float* probs = wsalloc((size_t)N_B * N_CAT);
  float* comb  = wsalloc((size_t)N_B * 1024);
  float* lab2  = wsalloc((size_t)N_B * 256);
  if (off * sizeof(float) > ws_size) return;  // workspace too small: bail

  float* out_features = (float*)d_out;                 // [64,512]
  float* out_cat      = out_features + N_B * 512;      // [64,32]
  float* out_lab      = out_cat + N_B * N_CAT;         // [64,512]

  auto gemm = [&](const float* A, const float* W, const float* bias,
                  const float* resid, float* C, int M, int N, int K,
                  int act, int bkn) {
    dim3 grid((M + 127) / 128, (N + 127) / 128);
    bool aligned = ((M & 127) == 0) && ((N & 127) == 0) && (bkn == 0);
    if (aligned)
      gemm_wmma_kernel<true><<<grid, dim3(256), 0, stream>>>(
          A, W, bias, resid, C, M, N, K, act, bkn);
    else
      gemm_wmma_kernel<false><<<grid, dim3(256), 0, stream>>>(
          A, W, bias, resid, C, M, N, K, act, bkn);
  };

  // ---- positional encoding: h = paths + pe ----
  {
    long long n = (long long)MTOK * 512;
    add_pe_kernel<<<(unsigned)((n + 255) / 256), 256, 0, stream>>>(paths, h, n);
  }

  // ---- PathEncoder: 2 pre-norm transformer layers ----
  for (int L = 0; L < 2; ++L) {
    const float* qkvw = enc_qkv_w + (size_t)L * 1536 * 512;
    const float* qkvb = enc_qkv_b + (size_t)L * 1536;
    const float* outw = enc_out_w + (size_t)L * 512 * 512;
    const float* outb = enc_out_b + (size_t)L * 512;
    const float* ln1w = enc_ln1_w + (size_t)L * 512;
    const float* ln1b = enc_ln1_b + (size_t)L * 512;
    const float* ln2w = enc_ln2_w + (size_t)L * 512;
    const float* ln2b = enc_ln2_b + (size_t)L * 512;
    const float* ff1w = enc_ff1_w + (size_t)L * 2048 * 512;
    const float* ff1b = enc_ff1_b + (size_t)L * 2048;
    const float* ff2w = enc_ff2_w + (size_t)L * 512 * 2048;
    const float* ff2b = enc_ff2_b + (size_t)L * 512;

    // x = LN1(h)
    layernorm512_kernel<<<(unsigned)MTOK, 256, 0, stream>>>(h, 512, ln1w, ln1b,
                                                            xbuf, 512, 0);
    // qkv + attention, chunked over 512 paths (32768 token rows) at a time.
    for (int pc = 0; pc < 4; ++pc) {
      int p0 = pc * 512;
      gemm(xbuf + (size_t)p0 * 64 * 512, qkvw, qkvb, nullptr, big,
           32768, 1536, 512, 0, 0);
      path_attn_kernel<<<dim3(512 * 8), 256, 0, stream>>>(big, masks, xbuf, p0);
    }
    // h = h + attn_out @ Wout^T + bout
    gemm(xbuf, outw, outb, h, h, (int)MTOK, 512, 512, 0, 0);
    // x = LN2(h)
    layernorm512_kernel<<<(unsigned)MTOK, 256, 0, stream>>>(h, 512, ln2w, ln2b,
                                                            xbuf, 512, 0);
    // FFN, chunked over 32768 rows
    for (int c = 0; c < 4; ++c) {
      size_t r0 = (size_t)c * 32768;
      gemm(xbuf + r0 * 512, ff1w, ff1b, nullptr, big, 32768, 2048, 512, 1, 0);
      gemm(big, ff2w, ff2b, h + r0 * 512, h + r0 * 512, 32768, 512, 2048, 0, 0);
    }
  }

  // path_emb = LN(h[:,0,:])  (strided rows: L*D apart)
  layernorm512_kernel<<<N_PATH, 256, 0, stream>>>(h, (long long)L_SEQ * 512,
                                                  path_norm_w, path_norm_b,
                                                  path_emb, 512, 0);

  // v_enh = relu(LN(variants @ vp_w^T + vp_b))
  gemm(variants, vp_w, vp_b, nullptr, tmp1, N_B, 512, 512, 0, 0);
  layernorm512_kernel<<<N_B, 256, 0, stream>>>(tmp1, 512, vp_ln_w, vp_ln_b,
                                               venh, 512, 1);

  // aggregation q/k/v projections
  gemm(venh,     agg_qkv_w,                 agg_qkv_b,        nullptr, qagg, N_B,    512, 512, 0, 0);
  gemm(path_emb, agg_qkv_w + 512 * 512,     agg_qkv_b + 512,  nullptr, kagg, N_PATH, 512, 512, 0, 0);
  gemm(path_emb, agg_qkv_w + 2 * 512 * 512, agg_qkv_b + 1024, nullptr, vagg, N_PATH, 512, 512, 0, 0);

  // ragged masked cross-attention over paths
  agg_attn_kernel<<<N_B * N_HEADS, 256, 0, stream>>>(qagg, kagg, vagg, p2s,
                                                     aggheads);
  // agg = LN(aggheads @ Wo^T + bo + v_enh)
  gemm(aggheads, agg_out_w, agg_out_b, venh, tmp1, N_B, 512, 512, 0, 0);
  layernorm512_kernel<<<N_B, 256, 0, stream>>>(tmp1, 512, agg_norm_w,
                                               agg_norm_b, aggn, 512, 0);
  // features = has_path ? agg : v_enh   -> d_out[0:32768]
  select_features_kernel<<<(N_B * 512 + 255) / 256, 256, 0, stream>>>(
      aggn, venh, pps, out_features, N_B * 512);

  // category head: c = relu(LN(feat @ W1^T + b1)); cat_logits = c @ W2^T + b2
  gemm(out_features, cat_w1, cat_b1, nullptr, tmp1, N_B, 512, 512, 0, 0);
  layernorm512_kernel<<<N_B, 256, 0, stream>>>(tmp1, 512, cat_ln_w, cat_ln_b,
                                               ghid, 512, 1);
  gemm(ghid, cat_w2, cat_b2, nullptr, out_cat, N_B, N_CAT, 512, 0, 0);

  // hierarchical label head
  sigmoid_kernel<<<(N_B * N_CAT + 255) / 256, 256, 0, stream>>>(out_cat, probs,
                                                                N_B * N_CAT);
  gemm(probs, cat_emb, nullptr, nullptr, wce, N_B, 512, N_CAT, 0, /*b_kn=*/1);
  gemm(probs, gate_w1, gate_b1, nullptr, ghid, N_B, 512, N_CAT, 1, 0);
  gemm(ghid, gate_w2, gate_b2, nullptr, gbuf, N_B, 512, 512, 2, 0);  // sigmoid
  build_comb_kernel<<<(N_B * 512 + 255) / 256, 256, 0, stream>>>(
      out_features, gbuf, wce, comb, N_B * 512);
  gemm(comb, lab_w1, lab_b1, nullptr, tmp1, N_B, 512, 1024, 0, 0);
  layernorm512_kernel<<<N_B, 256, 0, stream>>>(tmp1, 512, lab_ln_w, lab_ln_b,
                                               lab1, 512, 1);
  gemm(lab1, lab_w2, lab_b2, nullptr, lab2, N_B, 256, 512, 1, 0);
  gemm(lab2, lab_w3, lab_b3, nullptr, out_lab, N_B, N_CLS, 256, 0, 0);
}
